// SeqDecoder_62019327754323
// MI455X (gfx1250) — compile-verified, matrix operand order checked by reading, not run on hardware
//
#include <hip/hip_runtime.h>
#include <hip/hip_bf16.h>

// ---------------------------------------------------------------------------
// SeqDecoder LSTM on gfx1250 via v_wmma_f32_16x16x32_bf16.
//
//   B=1024 batch, E=1024 hidden, T=10 steps, OUT=34.
//   t=0 : gates = h0 @ W_hh^T + (b_ih+b_hh)            (x is zero)
//   t>0 : gates = h  @ (W_ih+W_hh)^T + (b_ih+b_hh)     (x == h)
//   out  = preds @ W_out^T + b_out
// ---------------------------------------------------------------------------

typedef __attribute__((ext_vector_type(16))) __bf16 v16bf;
typedef __attribute__((ext_vector_type(8)))  float  v8f;

#define E_DIM 1024
#define B_DIM 1024
#define T_STEPS 10
#define OUT_DIM 34
#define OUT_PAD 48   // 3 x 16 N-tiles

struct Bf16Frag { uint4 lo; uint4 hi; };

// Per-lane WMMA 16-bit operand fragment (A 16x32 row-major / B as rows of W):
// lane L holds row (base + L%16), K-halfs [k0 .. k0+7] and [k0+16 .. k0+23]
// where k0 = kBase + (L>>4)*8.  Two aligned 16B loads.
static __device__ inline v16bf load_frag_bf16(const __bf16* __restrict__ base,
                                              int row, int k, int ld) {
    const __bf16* p = base + (size_t)row * ld + k;
    Bf16Frag f;
    f.lo = *reinterpret_cast<const uint4*>(p);
    f.hi = *reinterpret_cast<const uint4*>(p + 16);
    return __builtin_bit_cast(v16bf, f);
}

static __device__ inline float fast_sigmoid(float x) {
    return 1.0f / (1.0f + __expf(-x));
}
static __device__ inline float fast_tanh(float x) {
    float e = __expf(2.0f * x);
    return (e - 1.0f) / (e + 1.0f);
}

// ------------------------------- prep kernels ------------------------------

__global__ void prep_weights(const float* __restrict__ Wih,
                             const float* __restrict__ Whh,
                             __bf16* __restrict__ Whh_bf,
                             __bf16* __restrict__ Wc_bf) {
    int i = blockIdx.x * blockDim.x + threadIdx.x;   // 4E*E = 4M elements
    float wh = Whh[i];
    Whh_bf[i] = (__bf16)wh;
    Wc_bf[i]  = (__bf16)(Wih[i] + wh);
}

__global__ void prep_h0(const float* __restrict__ h0, __bf16* __restrict__ h0b) {
    int i = blockIdx.x * blockDim.x + threadIdx.x;   // B*E elements
    h0b[i] = (__bf16)h0[i];
}

__global__ void prep_bias(const float* __restrict__ bih,
                          const float* __restrict__ bhh,
                          float* __restrict__ bsum) {
    int i = blockIdx.x * blockDim.x + threadIdx.x;   // 4E elements
    bsum[i] = bih[i] + bhh[i];
}

__global__ void prep_wout(const float* __restrict__ Wout,
                          __bf16* __restrict__ WoutP) {
    int i = blockIdx.x * blockDim.x + threadIdx.x;   // OUT_PAD*E elements
    WoutP[i] = (i < OUT_DIM * E_DIM) ? (__bf16)Wout[i] : (__bf16)0.0f;
}

// ------------------------------- LSTM step ---------------------------------
// Wave tile: 32 rows (batch) x 16 cols (hidden), x 4 gates (N, N+E, N+2E, N+3E).
// 2048 waves = 32 row-pairs x 64 col-tiles; 8 waves / 256-thread block.

__global__ __launch_bounds__(256) void lstm_step(
    const __bf16* __restrict__ hA,      // [B,E] bf16, previous hidden
    const __bf16* __restrict__ W,       // [4E,E] bf16 (Whh for t=0, Wc else)
    const float*  __restrict__ bsum,    // [4E]
    const float*  __restrict__ c_in,    // [B,E]
    float*        __restrict__ c_out,   // [B,E]
    __bf16*       __restrict__ hOut,    // [B,E] bf16
    __bf16*       __restrict__ preds,   // [B*T,E] bf16, row = b*T + t
    int t) {
    const int lane = threadIdx.x & 31;
    const int wid  = blockIdx.x * (blockDim.x >> 5) + (threadIdx.x >> 5);
    const int mt   = wid >> 6;        // 0..31  -> rows [mt*32, mt*32+31]
    const int nt   = wid & 63;        // 0..63  -> cols [nt*16, nt*16+15]
    const int m0   = mt * 32;
    const int n0   = nt * 16;
    const int lrow = lane & 15;
    const int koff = (lane >> 4) << 3;

    v8f acc[2][4] = {};               // [m-subtile][gate i,f,g,o]

    for (int k = 0; k < E_DIM; k += 32) {
        v16bf a0 = load_frag_bf16(hA, m0 + lrow,      k + koff, E_DIM);
        v16bf a1 = load_frag_bf16(hA, m0 + 16 + lrow, k + koff, E_DIM);
#pragma unroll
        for (int gi = 0; gi < 4; ++gi) {
            v16bf b = load_frag_bf16(W, gi * E_DIM + n0 + lrow, k + koff, E_DIM);
            acc[0][gi] = __builtin_amdgcn_wmma_f32_16x16x32_bf16(
                false, a0, false, b, (short)0, acc[0][gi], false, false);
            acc[1][gi] = __builtin_amdgcn_wmma_f32_16x16x32_bf16(
                false, a1, false, b, (short)0, acc[1][gi], false, false);
        }
    }

    // Fused LSTM-cell epilogue. C/D layout: vgpr r, lane L -> element
    // (M = r + 8*(L>>4), N = L&15) of the 16x16 tile.
    const int ncol = n0 + lrow;
    const float bi = bsum[ncol];
    const float bf = bsum[E_DIM + ncol];
    const float bg = bsum[2 * E_DIM + ncol];
    const float bo = bsum[3 * E_DIM + ncol];
    const int mhi = (lane >> 4) << 3;

#pragma unroll
    for (int ms = 0; ms < 2; ++ms) {
#pragma unroll
        for (int r = 0; r < 8; ++r) {
            const int brow = m0 + ms * 16 + r + mhi;
            const size_t idx = (size_t)brow * E_DIM + ncol;
            float iv = fast_sigmoid(acc[ms][0][r] + bi);
            float fv = fast_sigmoid(acc[ms][1][r] + bf);
            float gv = fast_tanh  (acc[ms][2][r] + bg);
            float ov = fast_sigmoid(acc[ms][3][r] + bo);
            float cn = fv * c_in[idx] + iv * gv;
            float hn = ov * fast_tanh(cn);
            c_out[idx] = cn;
            __bf16 hb = (__bf16)hn;
            hOut[idx] = hb;
            preds[((size_t)brow * T_STEPS + t) * E_DIM + ncol] = hb;
        }
    }
}

// ------------------------------ projection ---------------------------------
// out[M=10240, 34] = preds[M,1024] @ WoutP[48,1024]^T + b_out.
// 640 M-tiles; each wave does one M-tile x 3 N-tiles.

__global__ __launch_bounds__(256) void proj_out(
    const __bf16* __restrict__ preds,   // [B*T, E]
    const __bf16* __restrict__ WoutP,   // [OUT_PAD, E]
    const float*  __restrict__ bout,    // [OUT]
    float*        __restrict__ out) {   // [B*T, OUT]
    const int lane = threadIdx.x & 31;
    const int wid  = blockIdx.x * (blockDim.x >> 5) + (threadIdx.x >> 5);
    const int m0   = wid * 16;
    const int lrow = lane & 15;
    const int koff = (lane >> 4) << 3;

    v8f acc[3] = {};
    for (int k = 0; k < E_DIM; k += 32) {
        v16bf a = load_frag_bf16(preds, m0 + lrow, k + koff, E_DIM);
#pragma unroll
        for (int nt = 0; nt < 3; ++nt) {
            v16bf b = load_frag_bf16(WoutP, nt * 16 + lrow, k + koff, E_DIM);
            acc[nt] = __builtin_amdgcn_wmma_f32_16x16x32_bf16(
                false, a, false, b, (short)0, acc[nt], false, false);
        }
    }

    const int mhi = (lane >> 4) << 3;
#pragma unroll
    for (int nt = 0; nt < 3; ++nt) {
        const int col = nt * 16 + lrow;
        if (col < OUT_DIM) {
            const float bb = bout[col];
#pragma unroll
            for (int r = 0; r < 8; ++r) {
                const int row = m0 + r + mhi;
                out[(size_t)row * OUT_DIM + col] = acc[nt][r] + bb;
            }
        }
    }
}

// ------------------------------- launcher ----------------------------------

extern "C" void kernel_launch(void* const* d_in, const int* in_sizes, int n_in,
                              void* d_out, int out_size, void* d_ws, size_t ws_size,
                              hipStream_t stream) {
    const float* h0   = (const float*)d_in[0];   // [B,E]
    const float* c0   = (const float*)d_in[1];   // [B,E]
    const float* Wih  = (const float*)d_in[2];   // [4E,E]
    const float* Whh  = (const float*)d_in[3];   // [4E,E]
    const float* bih  = (const float*)d_in[4];   // [4E]
    const float* bhh  = (const float*)d_in[5];   // [4E]
    const float* Wout = (const float*)d_in[6];   // [OUT,E]
    const float* bout = (const float*)d_in[7];   // [OUT]
    // d_in[8] = future_preds (device scalar, always 10; cannot sync-read
    // during graph capture -> fixed T_STEPS).

    // Workspace carve-up (256B aligned slabs).
    char* ws = (char*)d_ws;
    size_t off = 0;
    auto alloc = [&](size_t bytes) {
        void* p = ws + off;
        off += (bytes + 255) & ~(size_t)255;
        return p;
    };
    __bf16* Whh_bf = (__bf16*)alloc((size_t)4 * E_DIM * E_DIM * 2);   // 8 MB
    __bf16* Wc_bf  = (__bf16*)alloc((size_t)4 * E_DIM * E_DIM * 2);   // 8 MB
    __bf16* WoutP  = (__bf16*)alloc((size_t)OUT_PAD * E_DIM * 2);     // 96 KB
    float*  bsum   = (float*) alloc((size_t)4 * E_DIM * 4);           // 16 KB
    __bf16* hbuf0  = (__bf16*)alloc((size_t)B_DIM * E_DIM * 2);       // 2 MB
    __bf16* hbuf1  = (__bf16*)alloc((size_t)B_DIM * E_DIM * 2);       // 2 MB
    float*  cbuf   = (float*) alloc((size_t)B_DIM * E_DIM * 4);       // 4 MB
    __bf16* preds  = (__bf16*)alloc((size_t)B_DIM * T_STEPS * E_DIM * 2); // 20 MB
    (void)ws_size; (void)in_sizes; (void)n_in; (void)out_size;

    // --- prep ---
    prep_weights<<<(4 * E_DIM * E_DIM) / 256, 256, 0, stream>>>(Wih, Whh, Whh_bf, Wc_bf);
    prep_h0     <<<(B_DIM * E_DIM) / 256, 256, 0, stream>>>(h0, hbuf0);
    prep_bias   <<<(4 * E_DIM) / 256, 256, 0, stream>>>(bih, bhh, bsum);
    prep_wout   <<<(OUT_PAD * E_DIM) / 256, 256, 0, stream>>>(Wout, WoutP);

    // --- recurrent steps (per-step launch = grid-wide barrier) ---
    for (int t = 0; t < T_STEPS; ++t) {
        const __bf16* hA = (t & 1) ? hbuf1 : hbuf0;
        __bf16*     hOut = (t & 1) ? hbuf0 : hbuf1;
        const __bf16* W  = (t == 0) ? Whh_bf : Wc_bf;
        const float* cin = (t == 0) ? c0 : cbuf;
        lstm_step<<<256, 256, 0, stream>>>(hA, W, bsum, cin, cbuf, hOut, preds, t);
    }

    // --- output projection ---
    proj_out<<<(B_DIM * T_STEPS / 16) / 8, 256, 0, stream>>>(preds, WoutP, bout, (float*)d_out);
}
